// NormalizedCorrelation2D_12438225289752
// MI455X (gfx1250) — compile-verified, matrix operand order checked by reading, not run on hardware
//
#include <hip/hip_runtime.h>
#include <hip/hip_bf16.h>

#define BATCH 64
#define WIDTH 513
#define WPAD  528          // 33*16, zero-padded rows
#define CH    512
#define PADW  256
#define NTILE 33           // ceil(513/16)
#define LPITCH 528         // LDS row pitch in bf16 elements (1056B, 32B aligned)

typedef __attribute__((ext_vector_type(16))) __bf16         v16bf;
typedef __attribute__((ext_vector_type(8)))  __bf16         v8bf;
typedef __attribute__((ext_vector_type(16))) unsigned short v16u;
typedef __attribute__((ext_vector_type(8)))  unsigned short v8u;
typedef __attribute__((ext_vector_type(8)))  float          v8f;
typedef __attribute__((ext_vector_type(4)))  float          v4f;
typedef __attribute__((ext_vector_type(4)))  unsigned int   v4u;
typedef __attribute__((ext_vector_type(8)))  int            v8i;
typedef __attribute__((ext_vector_type(4)))  int            v4i;

#if defined(__has_builtin)
#if __has_builtin(__builtin_amdgcn_tensor_load_to_lds)
#define HAVE_TDM 1
#endif
#endif

// Manual f32 -> bf16 round-to-nearest-even (inputs are finite).
__device__ __forceinline__ unsigned short f32_to_bf16_rne(float f) {
    unsigned int u = __float_as_uint(f);
    u += 0x7fffu + ((u >> 16) & 1u);
    return (unsigned short)(u >> 16);
}
__device__ __forceinline__ float bf16_us_to_f32(unsigned short h) {
    return __uint_as_float(((unsigned int)h) << 16);
}

__global__ void __launch_bounds__(256)
zero_out_kernel(float* __restrict__ out, int n) {
    int i = blockIdx.x * 256 + threadIdx.x;
    if (i < n) out[i] = 0.0f;
}

// Per-(b,c) inverse L2 norms over the width axis, fused product inv1*inv2.
__global__ void __launch_bounds__(256)
norm_scale_kernel(const float* __restrict__ x1, const float* __restrict__ x2,
                  float* __restrict__ scale) {
    int idx = blockIdx.x * 256 + threadIdx.x;          // 0 .. 64*512-1
    if (idx >= BATCH * CH) return;
    int b = idx >> 9;
    int c = idx & (CH - 1);
    const float* p1 = x1 + (size_t)b * WIDTH * CH + c;
    const float* p2 = x2 + (size_t)b * WIDTH * CH + c;
    float s1 = 0.0f, s2 = 0.0f;
    for (int w = 0; w < WIDTH; ++w) {
        float a = p1[(size_t)w * CH];
        float d = p2[(size_t)w * CH];
        s1 = fmaf(a, a, s1);
        s2 = fmaf(d, d, s2);
    }
    s1 = fmaxf(s1, 1e-12f);
    s2 = fmaxf(s2, 1e-12f);
    scale[idx] = rsqrtf(s1) * rsqrtf(s2);
}

// ---------------------------------------------------------------------------
// FAST PATH: pack x1*scale and x2 into bf16 hi/lo arrays [B][WPAD][512] in ws.
// K channels are stored fragment-major per 32-block: {0..7,16..23,8..15,24..31}
// so a lane's whole 16-element WMMA operand fragment is one contiguous 32B.
// Applied identically to A and B => contraction result unchanged.
// ---------------------------------------------------------------------------
__global__ void __launch_bounds__(256)
pack_bf16_kernel(const float* __restrict__ x1, const float* __restrict__ x2,
                 const float* __restrict__ scale,
                 unsigned short* __restrict__ Ahi, unsigned short* __restrict__ Alo,
                 unsigned short* __restrict__ Bhi, unsigned short* __restrict__ Blo) {
    int t = blockIdx.x * 256 + threadIdx.x;       // one thread per 8-channel octet
    if (t >= BATCH * WPAD * 64) return;
    int o8  = t & 63;                             // destination octet in row
    int row = (t >> 6) % WPAD;                    // 0..527
    int b   = t / (64 * WPAD);
    int o0  = o8 * 8;
    int j0  = o0 & 31;                            // octet position within 32-block
    int src0 = (o0 & ~31) | ((j0 == 8) ? 16 : (j0 == 16) ? 8 : j0);  // fragment-major perm

    float va[8], vb[8];
    if (row < WIDTH) {
        const float* p1 = x1 + ((size_t)(b * WIDTH + row)) * CH + src0;
        const float* p2 = x2 + ((size_t)(b * WIDTH + row)) * CH + src0;
        const float* sc = scale + b * CH + src0;
        #pragma unroll
        for (int i = 0; i < 8; i++) { va[i] = p1[i] * sc[i]; vb[i] = p2[i]; }
    } else {
        #pragma unroll
        for (int i = 0; i < 8; i++) { va[i] = 0.0f; vb[i] = 0.0f; }
    }
    v8u ah, al, bh, bl;
    #pragma unroll
    for (int i = 0; i < 8; i++) {
        unsigned short h = f32_to_bf16_rne(va[i]);
        ah[i] = h;
        al[i] = f32_to_bf16_rne(va[i] - bf16_us_to_f32(h));
        h = f32_to_bf16_rne(vb[i]);
        bh[i] = h;
        bl[i] = f32_to_bf16_rne(vb[i] - bf16_us_to_f32(h));
    }
    size_t dst = ((size_t)(b * WPAD + row)) * CH + o0;
    *(v8u*)(Ahi + dst) = ah;
    *(v8u*)(Alo + dst) = al;
    *(v8u*)(Bhi + dst) = bh;
    *(v8u*)(Blo + dst) = bl;
}

#ifdef HAVE_TDM
// TDM descriptor: 2D tile 16 rows x 512 bf16 (1024B/row), contiguous in global,
// LDS destination padded +32B per row (pad_interval=256 DWORDs, pad_amount=8 DWORDs)
// => LDS row pitch 1056B = LPITCH elements.
__device__ __forceinline__ void tdm_load_tile_16x512(unsigned lds_off,
                                                     const unsigned short* gsrc) {
    unsigned long long ga = (unsigned long long)(const void*)gsrc;
    v4u g0 = { 1u,                                  // count=1, user descriptor
               lds_off,                             // lds_addr
               (unsigned)ga,                        // global_addr[31:0]
               (unsigned)((ga >> 32) & 0x1ffffffull) | 0x80000000u };  // ga[56:32] | type=2
    v8i g1 = { (int)((1u << 16) | (1u << 20) | (7u << 22) | (7u << 25)),
               //     data_size=2B | pad_en | pad_interval=256dw | pad_amount=8dw
               (int)(512u << 16),                   // tensor_dim0 = 512 (lo16 @ bit48)
               (int)(16u  << 16),                   // tensor_dim0 hi=0 | tensor_dim1 = 16
               (int)(512u << 16),                   // tensor_dim1 hi=0 | tile_dim0 = 512
               16,                                  // tile_dim1 = 16, tile_dim2 = 0
               512,                                 // tensor_dim0_stride lo32 = 512 elems
               0, 0 };
    v4i z4 = {0, 0, 0, 0};
#if __clang_major__ >= 23
    v8i z8 = {0, 0, 0, 0, 0, 0, 0, 0};
    __builtin_amdgcn_tensor_load_to_lds(g0, g1, z4, z4, z8, 0);
#else
    __builtin_amdgcn_tensor_load_to_lds(g0, g1, z4, z4, 0);
#endif
}
#endif

// One WG per (batch, 16-row M tile). A tile DMA'd to LDS via TDM; 8 waves split
// the 33 N tiles; inner loop is pure ds/global b128 loads + 3 compensated WMMAs.
__global__ void __launch_bounds__(256)
corr_wmma_packed_kernel(const unsigned short* __restrict__ Ahi,
                        const unsigned short* __restrict__ Alo,
                        const unsigned short* __restrict__ Bhi,
                        const unsigned short* __restrict__ Blo,
                        float* __restrict__ out) {
    __shared__ __align__(32) unsigned short lAhi[16 * LPITCH];
    __shared__ __align__(32) unsigned short lAlo[16 * LPITCH];
    __shared__ float cScr[8][16][17];
    __shared__ float accOut[WIDTH];

    const int b    = blockIdx.x;
    const int mt   = blockIdx.y;
    const int m0   = mt * 16;
    const int tid  = threadIdx.x;
    const int wave = tid >> 5;
    const int lane = tid & 31;
    const int r    = lane & 15;
    const int half = lane >> 4;

    const size_t aOff = ((size_t)(b * WPAD + m0)) * CH;
#ifdef HAVE_TDM
    if (wave == 0) {
        // Low 32 bits of a generic LDS pointer are the in-group LDS byte offset.
        tdm_load_tile_16x512((unsigned)(unsigned long long)(void*)lAhi, Ahi + aOff);
        tdm_load_tile_16x512((unsigned)(unsigned long long)(void*)lAlo, Alo + aOff);
#if __has_builtin(__builtin_amdgcn_s_wait_tensorcnt)
        __builtin_amdgcn_s_wait_tensorcnt(0);
#else
        asm volatile("s_wait_tensorcnt 0" ::: "memory");
#endif
    }
#else
    for (int idx = tid; idx < 16 * 64; idx += 256) {   // cooperative copy fallback
        int row = idx >> 6, c0 = (idx & 63) * 8;
        *(v8u*)&lAhi[row * LPITCH + c0] = *(const v8u*)(Ahi + aOff + (size_t)row * CH + c0);
        *(v8u*)&lAlo[row * LPITCH + c0] = *(const v8u*)(Alo + aOff + (size_t)row * CH + c0);
    }
#endif
    for (int i = tid; i < WIDTH; i += 256) accOut[i] = 0.0f;
    __syncthreads();

    for (int nt = wave; nt < NTILE; nt += 8) {
        v8f acc = {0.f, 0.f, 0.f, 0.f, 0.f, 0.f, 0.f, 0.f};
        const size_t brow = ((size_t)(b * WPAD + nt * 16 + r)) * CH + half * 16;
        const unsigned short* pbh = Bhi + brow;
        const unsigned short* pbl = Blo + brow;
        const unsigned short* pah = lAhi + r * LPITCH + half * 16;
        const unsigned short* pal = lAlo + r * LPITCH + half * 16;

        #pragma unroll 4
        for (int k0 = 0; k0 < CH; k0 += 32) {
            v16bf ah = *(const v16bf*)(pah + k0);   // 32B contiguous fragment
            v16bf al = *(const v16bf*)(pal + k0);
            v16bf bh = *(const v16bf*)(pbh + k0);
            v16bf bl = *(const v16bf*)(pbl + k0);
            acc = __builtin_amdgcn_wmma_f32_16x16x32_bf16(false, ah, false, bh,
                                                          (short)0, acc, false, false);
            acc = __builtin_amdgcn_wmma_f32_16x16x32_bf16(false, ah, false, bl,
                                                          (short)0, acc, false, false);
            acc = __builtin_amdgcn_wmma_f32_16x16x32_bf16(false, al, false, bh,
                                                          (short)0, acc, false, false);
        }

        // Spill C tile (VGPR v -> M = v + 8*half, N = r), reduce 31 wrapped diagonals.
        #pragma unroll
        for (int v = 0; v < 8; v++) cScr[wave][v + half * 8][r] = acc[v];
        if (lane < 31) {
            int d = lane - 15;
            float s = 0.0f;
            int mlo = d > 0 ? d : 0;
            int mhi = d < 0 ? 15 + d : 15;
            for (int m = mlo; m <= mhi; m++) s += cScr[wave][m][m - d];
            int sidx = (m0 - nt * 16 + PADW + d + 2 * WIDTH) % WIDTH;
            atomicAdd(&accOut[sidx], s);            // ds_add_f32
        }
    }
    __syncthreads();
    for (int i = tid; i < WIDTH; i += 256)
        atomicAdd(&out[(size_t)b * WIDTH + i], accOut[i]);  // global_atomic_add_f32
}

// ---------------------------------------------------------------------------
// FALLBACK PATH (small ws): round-1 kernel, converts f32->bf16 hi/lo in-kernel.
// ---------------------------------------------------------------------------
__global__ void __launch_bounds__(256)
corr_wmma_fallback_kernel(const float* __restrict__ x1, const float* __restrict__ x2,
                          const float* __restrict__ scale, float* __restrict__ out) {
    __shared__ __align__(16) unsigned short Fhi[16][520];
    __shared__ __align__(16) unsigned short Flo[16][520];
    __shared__ float cScr[8][16][17];
    __shared__ float accOut[WIDTH];

    const int b   = blockIdx.x;
    const int mt  = blockIdx.y;
    const int m0  = mt * 16;
    const int tid = threadIdx.x;

    for (int idx = tid; idx < 16 * CH; idx += 256) {
        int row = idx >> 9;
        int c   = idx & (CH - 1);
        int mg  = m0 + row;
        float v = 0.0f;
        if (mg < WIDTH)
            v = x1[((size_t)(b * WIDTH + mg)) * CH + c] * scale[b * CH + c];
        unsigned short h = f32_to_bf16_rne(v);
        Fhi[row][c] = h;
        Flo[row][c] = f32_to_bf16_rne(v - bf16_us_to_f32(h));
    }
    for (int i = tid; i < WIDTH; i += 256) accOut[i] = 0.0f;
    __syncthreads();

    const int wave = tid >> 5;
    const int lane = tid & 31;
    const int r    = lane & 15;
    const int half = lane >> 4;

    for (int nt = wave; nt < NTILE; nt += 8) {
        v8f acc = {0.f, 0.f, 0.f, 0.f, 0.f, 0.f, 0.f, 0.f};
        const int  ng     = nt * 16 + r;
        const bool bvalid = (ng < WIDTH);
        const float* brow = x2 + ((size_t)(b * WIDTH + (bvalid ? ng : 0))) * CH;

        for (int k0 = 0; k0 < CH; k0 += 32) {
            const int ka = k0 + half * 8;
            v8bf a0 = *(const v8bf*)&Fhi[r][ka];
            v8bf a1 = *(const v8bf*)&Fhi[r][ka + 16];
            v8bf l0 = *(const v8bf*)&Flo[r][ka];
            v8bf l1 = *(const v8bf*)&Flo[r][ka + 16];
            v16bf ahi = __builtin_shufflevector(a0, a1, 0,1,2,3,4,5,6,7,8,9,10,11,12,13,14,15);
            v16bf alo = __builtin_shufflevector(l0, l1, 0,1,2,3,4,5,6,7,8,9,10,11,12,13,14,15);

            v4f t0 = {0,0,0,0}, t1 = t0, t2 = t0, t3 = t0;
            if (bvalid) {
                const float* bp = brow + ka;
                t0 = *(const v4f*)(bp);
                t1 = *(const v4f*)(bp + 4);
                t2 = *(const v4f*)(bp + 16);
                t3 = *(const v4f*)(bp + 20);
            }
            float fb[16];
            #pragma unroll
            for (int i = 0; i < 4; i++) {
                fb[i] = t0[i]; fb[4 + i] = t1[i]; fb[8 + i] = t2[i]; fb[12 + i] = t3[i];
            }
            v16u hu, lu;
            #pragma unroll
            for (int i = 0; i < 16; i++) {
                unsigned short h = f32_to_bf16_rne(fb[i]);
                hu[i] = h;
                lu[i] = f32_to_bf16_rne(fb[i] - bf16_us_to_f32(h));
            }
            v16bf bhi = __builtin_bit_cast(v16bf, hu);
            v16bf blo = __builtin_bit_cast(v16bf, lu);

            acc = __builtin_amdgcn_wmma_f32_16x16x32_bf16(false, ahi, false, bhi,
                                                          (short)0, acc, false, false);
            acc = __builtin_amdgcn_wmma_f32_16x16x32_bf16(false, ahi, false, blo,
                                                          (short)0, acc, false, false);
            acc = __builtin_amdgcn_wmma_f32_16x16x32_bf16(false, alo, false, bhi,
                                                          (short)0, acc, false, false);
        }

        #pragma unroll
        for (int v = 0; v < 8; v++) cScr[wave][v + half * 8][r] = acc[v];
        if (lane < 31) {
            int d = lane - 15;
            float s = 0.0f;
            int mlo = d > 0 ? d : 0;
            int mhi = d < 0 ? 15 + d : 15;
            for (int m = mlo; m <= mhi; m++) s += cScr[wave][m][m - d];
            int sidx = (m0 - nt * 16 + PADW + d + 2 * WIDTH) % WIDTH;
            atomicAdd(&accOut[sidx], s);
        }
    }
    __syncthreads();
    for (int i = tid; i < WIDTH; i += 256)
        atomicAdd(&out[(size_t)b * WIDTH + i], accOut[i]);
}

extern "C" void kernel_launch(void* const* d_in, const int* in_sizes, int n_in,
                              void* d_out, int out_size, void* d_ws, size_t ws_size,
                              hipStream_t stream) {
    (void)in_sizes; (void)n_in; (void)out_size;
    const float* x1 = (const float*)d_in[0];
    const float* x2 = (const float*)d_in[1];
    float* out   = (float*)d_out;
    float* scale = (float*)d_ws;                       // 64*512 f32 = 128 KB

    zero_out_kernel<<<(BATCH * WIDTH + 255) / 256, 256, 0, stream>>>(out, BATCH * WIDTH);
    norm_scale_kernel<<<(BATCH * CH) / 256, 256, 0, stream>>>(x1, x2, scale);

    const size_t arrElems = (size_t)BATCH * WPAD * CH;     // 17,301,504
    const size_t scaleBytes = 131072;
    const size_t need = scaleBytes + 4 * arrElems * sizeof(unsigned short);  // ~138.6 MB
    dim3 grid(BATCH, NTILE);

    if (ws_size >= need) {
        unsigned short* Ahi = (unsigned short*)((char*)d_ws + scaleBytes);
        unsigned short* Alo = Ahi + arrElems;
        unsigned short* Bhi = Alo + arrElems;
        unsigned short* Blo = Bhi + arrElems;
        int packThreads = BATCH * WPAD * 64;
        pack_bf16_kernel<<<(packThreads + 255) / 256, 256, 0, stream>>>(
            x1, x2, scale, Ahi, Alo, Bhi, Blo);
        corr_wmma_packed_kernel<<<grid, 256, 0, stream>>>(Ahi, Alo, Bhi, Blo, out);
    } else {
        corr_wmma_fallback_kernel<<<grid, 256, 0, stream>>>(x1, x2, scale, out);
    }
}
